// DCNNormActBlock_31705448579244
// MI455X (gfx1250) — compile-verified, matrix operand order checked by reading, not run on hardware
//
#include <hip/hip_runtime.h>
#include <stdint.h>

typedef __attribute__((ext_vector_type(16))) __bf16 v16bf;
typedef __attribute__((ext_vector_type(8)))  __bf16 v8bf;
typedef __attribute__((ext_vector_type(4)))  __bf16 v4bf;
typedef __attribute__((ext_vector_type(8)))  float  v8f;
typedef int v4i_vs __attribute__((vector_size(4 * sizeof(int))));

#define B_   4
#define H_   64
#define W_   64
#define C_   256
#define F_   256
#define KTAP 9
#define KDIM (KTAP * C_)        /* 2304 */
#define NPIX (B_ * H_ * W_)     /* 16384 */
#define OFFSTRIDE 32
#define NCHUNK (KDIM / 32)      /* 72 K-chunks of 32 */

#if defined(__HIP_DEVICE_COMPILE__) && __has_builtin(__builtin_amdgcn_global_load_async_to_lds_b128)
#define ASYNC_OK 1
#else
#define ASYNC_OK 0
#endif

__device__ __forceinline__ v16bf cat16(v8bf lo, v8bf hi) {
  return __builtin_shufflevector(lo, hi, 0,1,2,3,4,5,6,7,8,9,10,11,12,13,14,15);
}

__device__ __forceinline__ unsigned short f2bf_bits(float f) {
  __bf16 h = (__bf16)f;
  return __builtin_bit_cast(unsigned short, h);
}

__device__ __forceinline__ float bf2f(unsigned int bits16) {
  return __builtin_bit_cast(float, bits16 << 16);
}

// 16-byte global -> LDS copy. Async (ASYNCcnt-tracked) on CDNA5 toolchains
// that expose the builtin; synchronous VGPR bounce otherwise.
__device__ __forceinline__ void copy_b128_to_lds(const unsigned short* g,
                                                 unsigned short* l) {
#if ASYNC_OK
  __builtin_amdgcn_global_load_async_to_lds_b128(
      (__attribute__((address_space(1))) v4i_vs*)(uintptr_t)g,
      (__attribute__((address_space(3))) v4i_vs*)(uintptr_t)l, 0, 0);
#else
  *(v8bf*)l = *(const v8bf*)g;
#endif
}

__device__ __forceinline__ void wait_async0() {
#if ASYNC_OK
#if __has_builtin(__builtin_amdgcn_s_wait_asynccnt)
  __builtin_amdgcn_s_wait_asynccnt(0);
#else
  asm volatile("s_wait_asynccnt 0x0" ::: "memory");
#endif
#endif
}

// ---------------------------------------------------------------------------
// Kernel 0: precision conversion + weight transposition.
//   xb   [NPIX][C]        bf16 copy of x (im2col + bilinear gather source)
//   wbT  [F=256][K=2304]  bf16 transposed dcn weights (contiguous B-frag cols)
//   owbT [32][K=2304]     bf16 transposed offset weights, padded 27 -> 32
// ---------------------------------------------------------------------------
__global__ void prep_kernel(const float* __restrict__ x,
                            const float* __restrict__ dcn_w,
                            const float* __restrict__ offset_w,
                            unsigned short* __restrict__ xb,
                            unsigned short* __restrict__ wbT,
                            unsigned short* __restrict__ owbT) {
  const int tid  = blockIdx.x * blockDim.x + threadIdx.x;
  const int nthr = gridDim.x * blockDim.x;
  for (int i = tid; i < NPIX * C_; i += nthr)
    xb[i] = f2bf_bits(x[i]);
  for (int i = tid; i < F_ * KDIM; i += nthr) {
    int f = i / KDIM, kk = i % KDIM;
    wbT[i] = f2bf_bits(dcn_w[kk * F_ + f]);
  }
  for (int i = tid; i < 32 * KDIM; i += nthr) {
    int n = i / KDIM, kk = i % KDIM;
    float v = (n < 27) ? offset_w[kk * 27 + n] : 0.0f;
    owbT[i] = f2bf_bits(v);
  }
}

// ---------------------------------------------------------------------------
// Kernel 1: offset 3x3 conv as WMMA GEMM. One wave per 16-pixel strip.
// M=16 pixels, N=32 (27 padded), K=2304, software-pipelined fragment loads.
// ---------------------------------------------------------------------------
__global__ void __launch_bounds__(256) offconv_kernel(
    const unsigned short* __restrict__ xb,
    const unsigned short* __restrict__ owbT,
    float* __restrict__ off) {
  const int lane = threadIdx.x & 31;
  const int wave = threadIdx.x >> 5;
  const int tile = blockIdx.x * 8 + wave;    // 0..1023
  const int p0   = tile * 16;
  const int w0   = p0 & 63;
  const int h    = (p0 >> 6) & 63;
  const int b    = p0 >> 12;
  const int M    = lane & 15;
  const int hi8  = (lane >> 4) << 3;

  v8f acc0 = {};
  v8f acc1 = {};
  const v8bf zz = {};

  auto loadA = [&](int kk) -> v16bf {
    const int k  = kk >> 8;
    const int ky = k / 3 - 1;
    const int kx = k % 3 - 1;
    const int y  = h + ky;
    const int xw = w0 + M + kx;
    v8bf alo = zz, ahi = zz;
    if ((y >= 0) && (y < H_) && (xw >= 0) && (xw < W_)) {
      const unsigned short* ap =
          xb + ((((b * H_ + y) * W_ + xw) * C_) + (kk & 255) + hi8);
      alo = *(const v8bf*)ap;
      ahi = *(const v8bf*)(ap + 16);
    }
    return cat16(alo, ahi);
  };
  auto loadB = [&](int n0, int kk) -> v16bf {
    const unsigned short* bp = owbT + ((n0 + M) * KDIM + kk + hi8);
    return cat16(*(const v8bf*)bp, *(const v8bf*)(bp + 16));
  };

  v16bf a  = loadA(0);
  v16bf b0 = loadB(0, 0);
  v16bf b1 = loadB(16, 0);
  for (int kk = 0; kk < KDIM; kk += 32) {
    v16bf an = a, b0n = b0, b1n = b1;
    const int nx = kk + 32;
    if (nx < KDIM) {                 // prefetch next chunk before the MACs
      an  = loadA(nx);
      b0n = loadB(0, nx);
      b1n = loadB(16, nx);
    }
    acc0 = __builtin_amdgcn_wmma_f32_16x16x32_bf16(false, a, false, b0,
                                                   (short)0, acc0, false, false);
    acc1 = __builtin_amdgcn_wmma_f32_16x16x32_bf16(false, a, false, b1,
                                                   (short)0, acc1, false, false);
    a = an; b0 = b0n; b1 = b1n;
  }

#pragma unroll
  for (int r = 0; r < 8; ++r) {
    const int p = p0 + r + hi8;
    off[p * OFFSTRIDE + M] = acc0[r];            // f = 0..15
    float v = acc1[r];                           // f = 16..31
    if (16 + M >= 18) v = 1.0f / (1.0f + __expf(-v));  // sigmoid mask lanes
    off[p * OFFSTRIDE + 16 + M] = v;
  }
}

// ---------------------------------------------------------------------------
// Kernel 2: fused deformable sampling + main GEMM + BN + ReLU.
// Block = 64 pixels x 256 outputs; 8 waves = 2(M-row) x 4(N-quarter),
// each wave owns 2 M-subtiles x 4 N-subtiles (8 WMMAs per 32-K step).
// B tiles are async-staged into double-buffered LDS one step ahead.
// ---------------------------------------------------------------------------
__global__ void __launch_bounds__(256) dcn_kernel(
    const unsigned short* __restrict__ xb,
    const float* __restrict__ off,
    const unsigned short* __restrict__ wbT,
    const float* __restrict__ gamma,
    const float* __restrict__ beta,
    const float* __restrict__ mmean,
    const float* __restrict__ mvar,
    float* __restrict__ out) {
  __shared__ __align__(16) unsigned short sA[64 * C_];      // 32 KB sampled A
  __shared__ __align__(16) unsigned short sB[2 * 256 * 32]; // 2 x 16 KB B tiles

  const int tid  = threadIdx.x;
  const int lane = tid & 31;
  const int wave = tid >> 5;
  const int mrow = wave >> 2;          // 0/1
  const int wcol = wave & 3;           // 64-wide N quarter
  const int M    = lane & 15;
  const int hi8  = (lane >> 4) << 3;

  const int pbase = blockIdx.x * 64;

  // sampling assignment: 4 threads per pixel, 4 channels per iteration each
  const int spix = tid >> 2;           // 0..63
  const int g    = tid & 3;            // 0..3
  const int P    = pbase + spix;
  const int ph   = (P >> 6) & 63;
  const int pw   = P & 63;
  const int pb   = P >> 12;

  v8f acc[2][4] = {};

  const int kyA[9] = {-1,-1,-1, 0,0,0, 1,1,1};
  const int kxA[9] = {-1, 0, 1,-1,0,1,-1,0,1};

  // stage one 32K x 256N B tile (16 KB): thread t copies row f = t
  auto stage_B = [&](int q, int buf) {
    const unsigned short* src = wbT + (size_t)tid * KDIM + q * 32;
    unsigned short* dst = sB + buf * 8192 + tid * 32;
#pragma unroll
    for (int c = 0; c < 4; ++c)
      copy_b128_to_lds(src + c * 8, dst + c * 8);
  };

  stage_B(0, 0);   // prologue: first chunk in flight during first sampling

  for (int k = 0; k < KTAP; ++k) {
    __syncthreads();   // previous tap's readers are done with sA

    // ---- bilinear-sample tap k (bf16 source) into sA ----
    const float dy = off[P * OFFSTRIDE + k];
    const float dx = off[P * OFFSTRIDE + 9 + k];
    const float mk = off[P * OFFSTRIDE + 18 + k];
    const float py = (float)(ph + kyA[k]) + dy;
    const float px = (float)(pw + kxA[k]) + dx;
    const float fy = floorf(py), fx = floorf(px);
    const float wy1 = py - fy, wx1 = px - fx;
    const float wy0 = 1.0f - wy1, wx0 = 1.0f - wx1;
    const int y0 = (int)fy, x0 = (int)fx;
    const int y1 = y0 + 1,  x1 = x0 + 1;
    const bool vy0 = (y0 >= 0) & (y0 < H_);
    const bool vy1 = (y1 >= 0) & (y1 < H_);
    const bool vx0 = (x0 >= 0) & (x0 < W_);
    const bool vx1 = (x1 >= 0) & (x1 < W_);
    const float w00 = wy0 * wx0 * mk * (float)(vy0 && vx0);
    const float w01 = wy0 * wx1 * mk * (float)(vy0 && vx1);
    const float w10 = wy1 * wx0 * mk * (float)(vy1 && vx0);
    const float w11 = wy1 * wx1 * mk * (float)(vy1 && vx1);
    const int y0c = min(max(y0, 0), H_ - 1), y1c = min(max(y1, 0), H_ - 1);
    const int x0c = min(max(x0, 0), W_ - 1), x1c = min(max(x1, 0), W_ - 1);
    const unsigned short* q00 = xb + (((pb * H_ + y0c) * W_ + x0c) * C_);
    const unsigned short* q01 = xb + (((pb * H_ + y0c) * W_ + x1c) * C_);
    const unsigned short* q10 = xb + (((pb * H_ + y1c) * W_ + x0c) * C_);
    const unsigned short* q11 = xb + (((pb * H_ + y1c) * W_ + x1c) * C_);

#pragma unroll
    for (int it = 0; it < 16; ++it) {
      const int c = it * 16 + g * 4;
      const uint2 r00 = *(const uint2*)(q00 + c);
      const uint2 r01 = *(const uint2*)(q01 + c);
      const uint2 r10 = *(const uint2*)(q10 + c);
      const uint2 r11 = *(const uint2*)(q11 + c);
      v4bf pk;
#pragma unroll
      for (int e = 0; e < 4; ++e) {
        const unsigned u00 = (e < 2) ? r00.x : r00.y;
        const unsigned u01 = (e < 2) ? r01.x : r01.y;
        const unsigned u10 = (e < 2) ? r10.x : r10.y;
        const unsigned u11 = (e < 2) ? r11.x : r11.y;
        const int sh = (e & 1) * 16;
        const float s = w00 * bf2f((u00 >> sh) & 0xffffu)
                      + w01 * bf2f((u01 >> sh) & 0xffffu)
                      + w10 * bf2f((u10 >> sh) & 0xffffu)
                      + w11 * bf2f((u11 >> sh) & 0xffffu);
        pk[e] = (__bf16)s;
      }
      *(v4bf*)(sA + spix * C_ + c) = pk;
    }

    // ---- 8 K-steps of WMMA for this tap, B double-buffered via async ----
#pragma unroll
    for (int s = 0; s < 8; ++s) {
      const int q = k * 8 + s;
      wait_async0();        // this wave's stage of chunk q has landed
      __syncthreads();      // all waves' stages landed; prev-step readers done
      if (q + 1 < NCHUNK) stage_B(q + 1, (q + 1) & 1);  // overlap next chunk

      const int c0 = s * 32;
      const unsigned short* ar0 = sA + (mrow * 16 + M) * C_ + c0 + hi8;
      const unsigned short* ar1 = sA + (32 + mrow * 16 + M) * C_ + c0 + hi8;
      const v16bf a0 = cat16(*(const v8bf*)ar0, *(const v8bf*)(ar0 + 16));
      const v16bf a1 = cat16(*(const v8bf*)ar1, *(const v8bf*)(ar1 + 16));
      const unsigned short* bbuf = sB + (q & 1) * 8192;
#pragma unroll
      for (int j = 0; j < 4; ++j) {
        const int f = wcol * 64 + j * 16 + M;
        const unsigned short* bp = bbuf + f * 32 + hi8;
        const v16bf bfr = cat16(*(const v8bf*)bp, *(const v8bf*)(bp + 16));
        acc[0][j] = __builtin_amdgcn_wmma_f32_16x16x32_bf16(
            false, a0, false, bfr, (short)0, acc[0][j], false, false);
        acc[1][j] = __builtin_amdgcn_wmma_f32_16x16x32_bf16(
            false, a1, false, bfr, (short)0, acc[1][j], false, false);
      }
    }
  }

  // ---- epilogue: inference BN + ReLU, coalesced stores ----
#pragma unroll
  for (int j = 0; j < 4; ++j) {
    const int f = wcol * 64 + j * 16 + M;
    const float inv = gamma[f] * rsqrtf(mvar[f] + 1e-3f);
    const float mu  = mmean[f];
    const float bt  = beta[f];
#pragma unroll
    for (int mi = 0; mi < 2; ++mi) {
#pragma unroll
      for (int r = 0; r < 8; ++r) {
        const int Pp = pbase + mi * 32 + mrow * 16 + r + hi8;
        const float v = (acc[mi][j][r] - mu) * inv + bt;
        out[Pp * F_ + f] = fmaxf(v, 0.0f);
      }
    }
  }
}

// ---------------------------------------------------------------------------
extern "C" void kernel_launch(void* const* d_in, const int* in_sizes, int n_in,
                              void* d_out, int out_size, void* d_ws, size_t ws_size,
                              hipStream_t stream) {
  (void)in_sizes; (void)n_in; (void)out_size; (void)ws_size;

  const float* x        = (const float*)d_in[0];
  const float* offset_w = (const float*)d_in[1];
  const float* dcn_w    = (const float*)d_in[2];
  const float* gamma    = (const float*)d_in[3];
  const float* beta     = (const float*)d_in[4];
  const float* mmean    = (const float*)d_in[5];
  const float* mvar     = (const float*)d_in[6];
  float* out = (float*)d_out;

  char* ws = (char*)d_ws;
  unsigned short* xb   = (unsigned short*)ws;  ws += (size_t)NPIX * C_ * 2;   // 8 MB
  unsigned short* wbT  = (unsigned short*)ws;  ws += (size_t)F_ * KDIM * 2;   // 1.2 MB
  unsigned short* owbT = (unsigned short*)ws;  ws += (size_t)32 * KDIM * 2;   // 144 KB
  float*          off  = (float*)ws;                                          // 2 MB

  prep_kernel<<<1024, 256, 0, stream>>>(x, dcn_w, offset_w, xb, wbT, owbT);
  offconv_kernel<<<128, 256, 0, stream>>>(xb, owbT, off);
  dcn_kernel<<<256, 256, 0, stream>>>(xb, off, wbT, gamma, beta, mmean, mvar, out);
}
